// DenoisingAutoEncoder_18545668784860
// MI455X (gfx1250) — matrix operand order repaired and checked by reading, not compiled, not run on hardware
//
#include <hip/hip_runtime.h>
#include <hip/hip_bf16.h>
#include <math.h>

#define DINLINE __device__ __forceinline__

typedef __attribute__((ext_vector_type(16))) __bf16        v16bf;
typedef __attribute__((ext_vector_type(8)))  float         v8f;
typedef __attribute__((ext_vector_type(8)))  unsigned short u16x8;

static constexpr int V_  = 32000;
static constexpr int E_  = 512;
static constexpr int H_  = 1024;   // hidden (== L1 == L2)
static constexpr int G4_ = 4096;   // 4*H (gate rows)
static constexpr int TX_ = 2048;
static constexpr int TY_ = 512;

// ---------------- bf16 helpers (round-to-nearest-even) ----------------
DINLINE unsigned short f2bf(float f) {
  unsigned u = __float_as_uint(f);
  u += 0x7FFFu + ((u >> 16) & 1u);
  return (unsigned short)(u >> 16);
}
DINLINE float bf2f(unsigned short h) { return __uint_as_float(((unsigned)h) << 16); }
DINLINE float sigm(float x) { return 1.0f / (1.0f + __expf(-x)); }

// vectorized bf16-row dot against LDS-staged f32 vector: len multiple of 8
DINLINE float dot_bf16_row(const unsigned short* __restrict__ wrow,
                           const float* __restrict__ hs, int len) {
  float acc = 0.f;
  const u16x8* wv = (const u16x8*)wrow;            // rows are 16B-aligned
  const int nc = len >> 3;
  for (int c = 0; c < nc; ++c) {
    u16x8 wkt = wv[c];                             // global_load_b128: 8 bf16/lane
    const float* hp = hs + (c << 3);
#pragma unroll
    for (int j = 0; j < 8; ++j) acc += bf2f(wkt[j]) * hp[j];
  }
  return acc;
}

// ---------------- device-wide barrier (persistent kernels) ----------------
DINLINE void grid_sync(volatile unsigned* cnt, volatile unsigned* gen, unsigned nblk) {
  __syncthreads();
  __threadfence();
  if (threadIdx.x == 0) {
    unsigned g = *gen;
    if (atomicAdd((unsigned*)cnt, 1u) == nblk - 1u) {
      *cnt = 0u;
      __threadfence();
      atomicAdd((unsigned*)gen, 1u);
    } else {
      while (*gen == g) { __builtin_amdgcn_s_sleep(2); }
    }
  }
  __syncthreads();
}

// ---------------- utility kernels ----------------
__global__ void k_zero_u32(unsigned* p, int n) {
  int i = blockIdx.x * blockDim.x + threadIdx.x;
  if (i < n) p[i] = 0u;
}

__global__ void k_cvt(const float* __restrict__ in, unsigned short* __restrict__ out, int n) {
  int i = blockIdx.x * blockDim.x + threadIdx.x;
  if (i < n) out[i] = f2bf(in[i]);
}

__global__ void k_embed(const int* __restrict__ x, const float* __restrict__ emb,
                        unsigned short* __restrict__ xe) {
  int i = blockIdx.x * blockDim.x + threadIdx.x;   // exactly TX*E threads
  int t = i >> 9;                                   // E_ == 512
  int e = i & (E_ - 1);
  xe[i] = f2bf(emb[(size_t)x[t] * E_ + e]);
}

__global__ void k_row0(float* __restrict__ out) {
  int i = blockIdx.x * blockDim.x + threadIdx.x;
  if (i < V_) out[i] = -logf((float)V_);            // log_softmax(zeros)
}

// ---------------- WMMA bf16 GEMM:  C[M,N] = A[M,K] * W[N,K]^T ----------------
// 256 threads = 8 waves; each wave owns one 16x16 C tile; block tile = 16M x 128N.
// Fragment k order: contiguous k = 16*hi + s for A and B (16-bit A 16x32; B 32x16).
__global__ void __launch_bounds__(256) k_gemm_bf16(
    const unsigned short* __restrict__ A,   // (M,K) bf16 row-major
    const unsigned short* __restrict__ W,   // (N,K) bf16 row-major
    float* __restrict__ Cout,               // (M,N) f32
    int M, int N, int K)
{
  const int lane = threadIdx.x & 31;
  const int wave = threadIdx.x >> 5;
  const int ln16 = lane & 15;
  const int hi   = lane >> 4;                       // 0: lanes 0-15, 1: lanes 16-31
  const int m0   = blockIdx.y * 16;
  const int n    = blockIdx.x * 128 + wave * 16 + ln16;
  const unsigned short* arow = A + (size_t)(m0 + ln16) * K + hi * 16;  // A K-offset 0 / 16
  const unsigned short* brow = W + (size_t)n * K + hi * 16;            // B K-offset 0 / 16

  v8f acc = {};
  for (int k0 = 0; k0 < K; k0 += 32) {
    union { v16bf v; struct { u16x8 lo, hi; } p; } af;
    af.p.lo = *(const u16x8*)(arow + k0);           // K = ka .. ka+7
    af.p.hi = *(const u16x8*)(arow + k0 + 8);       // K = ka+8 .. ka+15
    v16bf bv = *(const v16bf*)(const void*)(brow + k0);  // K = kb .. kb+15
    acc = __builtin_amdgcn_wmma_f32_16x16x32_bf16(false, af.v, false, bv,
                                                  (short)0, acc, false, false);
  }
  // C/D layout: lanes 0-15 rows r, lanes 16-31 rows 8+r; col = lane&15
  float* cp = Cout + (size_t)m0 * N + n;
#pragma unroll
  for (int r = 0; r < 8; ++r) {
    int mr = hi ? (8 + r) : r;
    cp[(size_t)mr * N] = acc[r];
  }
}

// ---------------- persistent sequential LSTM layer ----------------
// grid = 32 blocks x 128 threads = 4096 threads, one gate-row each.
// gates = G[t] (precomputed Wih@x via WMMA GEMM) + b + Whh[row,:] . h
// Row-major bf16 weights: each lane streams its row with b128 loads.
__global__ void __launch_bounds__(128) k_lstm_seq(
    const float* __restrict__ G,              // (T, 4H) input pre-projection
    const unsigned short* __restrict__ Whh,   // (4H, H) bf16 row-major
    const float* __restrict__ b,              // (4H)
    float* __restrict__ h,                    // (H) state in ws
    float* __restrict__ gates,                // (4H) scratch
    unsigned short* __restrict__ ys,          // (T,H) bf16 output seq or nullptr
    unsigned* sync_cnt, unsigned* sync_gen,
    int T)
{
  __shared__ float hs[H_];
  const int row  = blockIdx.x * blockDim.x + threadIdx.x;   // 0..4095
  const unsigned nblk = gridDim.x;
  float c_reg = 0.f;
  if (row < H_) h[row] = 0.f;
  grid_sync(sync_cnt, sync_gen, nblk);
  const float bias = b[row];
  const unsigned short* wrow = Whh + (size_t)row * H_;

  for (int t = 0; t < T; ++t) {
    for (int k = threadIdx.x; k < H_; k += blockDim.x) hs[k] = h[k];
    __syncthreads();
    float acc = G[(size_t)t * G4_ + row] + bias + dot_bf16_row(wrow, hs, H_);
    gates[row] = acc;
    grid_sync(sync_cnt, sync_gen, nblk);
    if (row < H_) {
      float ig = sigm(gates[row]);
      float fg = sigm(gates[H_ + row]);
      float gg = tanhf(gates[2 * H_ + row]);
      float og = sigm(gates[3 * H_ + row]);
      c_reg = fg * c_reg + ig * gg;
      float hn = og * tanhf(c_reg);
      h[row] = hn;
      if (ys) ys[(size_t)t * H_ + row] = f2bf(hn);
    }
    grid_sync(sync_cnt, sync_gen, nblk);
  }
}

// ---------------- FC bottleneck: f2 = relu(fc2 relu(fc1 h + b1) + b2) ----------------
__global__ void __launch_bounds__(1024) k_fc(
    const float* __restrict__ fc1_w, const float* __restrict__ fc1_b,
    const float* __restrict__ fc2_w, const float* __restrict__ fc2_b,
    const float* __restrict__ h_enc,
    float* __restrict__ h0, float* __restrict__ c0,
    float* __restrict__ h1, float* __restrict__ c1)
{
  __shared__ float f1s[H_];
  const int i = threadIdx.x;
  float a = fc1_b[i];
  const float* w1 = fc1_w + (size_t)i * H_;
  for (int k = 0; k < H_; ++k) a += w1[k] * h_enc[k];
  f1s[i] = fmaxf(a, 0.f);
  __syncthreads();
  float a2 = fc2_b[i];
  const float* w2 = fc2_w + (size_t)i * H_;
  for (int k = 0; k < H_; ++k) a2 += w2[k] * f1s[k];
  float f2v = fmaxf(a2, 0.f);
  h0[i] = f2v; c0[i] = f2v; h1[i] = f2v; c1[i] = f2v;
}

// ---------------- persistent autoregressive decoder ----------------
// 64 blocks x 256 threads = 16384 threads. Cell gates: row = gtid>>2, 4 K-partials
// reduced with lane shuffles. Vocab projection + fused online-softmax + argmax.
// All weights row-major bf16, streamed per-lane with b128 loads.
__global__ void __launch_bounds__(256) k_decoder(
    const int* __restrict__ y,
    const float* __restrict__ dWih0,           // (4H,1) f32 column
    const float* __restrict__ db0, const float* __restrict__ db1,
    const unsigned short* __restrict__ dWhh0,  // (4H,H) bf16 row-major
    const unsigned short* __restrict__ dWih1,  // (4H,H)
    const unsigned short* __restrict__ dWhh1,  // (4H,H)
    const unsigned short* __restrict__ outW,   // (V,H) bf16 row-major
    const float* __restrict__ out_b,
    float* __restrict__ h0, float* __restrict__ c0i,
    float* __restrict__ h1, float* __restrict__ c1i,
    float* __restrict__ gates,                 // (4H)
    float* __restrict__ red, int* __restrict__ redi,  // per-block reductions
    float* __restrict__ pub,                   // [0]=logZ [1]=next token (float)
    unsigned* sync_cnt, unsigned* sync_gen,
    float* __restrict__ out)                   // (TY,V)
{
  __shared__ float hs0[H_];
  __shared__ float hs1[H_];
  __shared__ float rm[256], rs[256], rbv[256];
  __shared__ int   rbi[256];

  const int tid  = threadIdx.x;
  const int blk  = blockIdx.x;
  const unsigned nblk = gridDim.x;
  const int gtid = blk * blockDim.x + tid;          // 0..16383
  const int NT   = (int)nblk * blockDim.x;
  const int row  = gtid >> 2;                       // gate row 0..4095
  const int part = gtid & 3;                        // K partial 0..3

  float c0r = (gtid < H_) ? c0i[gtid] : 0.f;
  float c1r = (gtid < H_) ? c1i[gtid] : 0.f;
  if (gtid == 0) pub[1] = (float)y[0];
  grid_sync(sync_cnt, sync_gen, nblk);

  for (int s = 0; s < TY_ - 1; ++s) {
    const float xt = pub[1];

    // ---- cell 0: gates = b0 + Wih0*xt + Whh0 h0 ----
    for (int k = tid; k < H_; k += blockDim.x) hs0[k] = h0[k];
    __syncthreads();
    float acc = (part == 0) ? (db0[row] + dWih0[row] * xt) : 0.f;
    {
      const int kb = part * 256;                    // 256-wide K partial
      acc += dot_bf16_row(dWhh0 + (size_t)row * H_ + kb, hs0 + kb, 256);
    }
    acc += __shfl_xor(acc, 1);
    acc += __shfl_xor(acc, 2);
    if (part == 0) gates[row] = acc;
    grid_sync(sync_cnt, sync_gen, nblk);
    if (gtid < H_) {
      float ig = sigm(gates[gtid]);
      float fg = sigm(gates[H_ + gtid]);
      float gg = tanhf(gates[2 * H_ + gtid]);
      float og = sigm(gates[3 * H_ + gtid]);
      c0r = fg * c0r + ig * gg;
      h0[gtid] = og * tanhf(c0r);
    }
    grid_sync(sync_cnt, sync_gen, nblk);

    // ---- cell 1: gates = b1 + Wih1 h0 + Whh1 h1 ----
    for (int k = tid; k < H_; k += blockDim.x) { hs0[k] = h0[k]; hs1[k] = h1[k]; }
    __syncthreads();
    acc = (part == 0) ? db1[row] : 0.f;
    {
      const unsigned short* Wp = (part < 2) ? dWih1 : dWhh1;
      const float* hp = (part < 2) ? hs0 : hs1;
      const int kb = (part & 1) * 512;              // 512-wide K partial
      acc += dot_bf16_row(Wp + (size_t)row * H_ + kb, hp + kb, 512);
    }
    acc += __shfl_xor(acc, 1);
    acc += __shfl_xor(acc, 2);
    if (part == 0) gates[row] = acc;
    grid_sync(sync_cnt, sync_gen, nblk);
    if (gtid < H_) {
      float ig = sigm(gates[gtid]);
      float fg = sigm(gates[H_ + gtid]);
      float gg = tanhf(gates[2 * H_ + gtid]);
      float og = sigm(gates[3 * H_ + gtid]);
      c1r = fg * c1r + ig * gg;
      h1[gtid] = og * tanhf(c1r);
    }
    grid_sync(sync_cnt, sync_gen, nblk);

    // ---- logits = out_w h1 + out_b; fused online-softmax + argmax ----
    for (int k = tid; k < H_; k += blockDim.x) hs1[k] = h1[k];
    __syncthreads();
    const int v0 = gtid, v1 = gtid + NT;            // v0 < 16384, v1 may exceed V
    float l0, l1 = 0.f;
    l0 = out_b[v0] + dot_bf16_row(outW + (size_t)v0 * H_, hs1, H_);
    float m = l0, ssum = 1.f, bestv = l0;
    int besti = v0;
    if (v1 < V_) {
      float a = out_b[v1] + dot_bf16_row(outW + (size_t)v1 * H_, hs1, H_);
      l1 = a;
      if (a > m) { ssum = ssum * __expf(m - a) + 1.f; m = a; }
      else       { ssum += __expf(a - m); }
      if (a > bestv) { bestv = a; besti = v1; }     // ties keep smaller index
    }
    rm[tid] = m; rs[tid] = ssum; rbv[tid] = bestv; rbi[tid] = besti;
    __syncthreads();
    for (int off = 128; off > 0; off >>= 1) {
      if (tid < off) {
        float m2 = rm[tid + off], s2 = rs[tid + off];
        if (m2 > rm[tid]) { rs[tid] = s2 + rs[tid] * __expf(rm[tid] - m2); rm[tid] = m2; }
        else              { rs[tid] += s2 * __expf(m2 - rm[tid]); }
        float bv2 = rbv[tid + off]; int bi2 = rbi[tid + off];
        if (bv2 > rbv[tid] || (bv2 == rbv[tid] && bi2 < rbi[tid])) { rbv[tid] = bv2; rbi[tid] = bi2; }
      }
      __syncthreads();
    }
    if (tid == 0) {
      red[blk * 3 + 0] = rm[0];
      red[blk * 3 + 1] = rs[0];
      red[blk * 3 + 2] = rbv[0];
      redi[blk] = rbi[0];
    }
    grid_sync(sync_cnt, sync_gen, nblk);
    if (gtid == 0) {
      float M = -INFINITY, S = 0.f, BV = -INFINITY; int BI = 0;
      for (int q = 0; q < (int)nblk; ++q) {
        float m2 = red[q * 3], s2 = red[q * 3 + 1];
        if (m2 > M) { S = s2 + S * __expf(M - m2); M = m2; }
        else        { S += s2 * __expf(m2 - M); }
        float bv = red[q * 3 + 2]; int bi = redi[q];
        if (bv > BV || (bv == BV && bi < BI)) { BV = bv; BI = bi; }
      }
      pub[0] = M + __logf(S);
      pub[1] = (float)BI;                           // greedy feedback token
    }
    grid_sync(sync_cnt, sync_gen, nblk);
    const float lz = pub[0];
    float* orow = out + (size_t)(s + 1) * V_;
    orow[v0] = l0 - lz;
    if (v1 < V_) orow[v1] = l1 - lz;
  }
}

// ---------------- host launcher ----------------
extern "C" void kernel_launch(void* const* d_in, const int* in_sizes, int n_in,
                              void* d_out, int out_size, void* d_ws, size_t ws_size,
                              hipStream_t stream) {
  (void)in_sizes; (void)n_in; (void)out_size; (void)ws_size;
  const int*   x        = (const int*)d_in[0];
  const int*   y        = (const int*)d_in[1];
  const float* emb      = (const float*)d_in[2];
  const float* eWih0    = (const float*)d_in[3];
  const float* eWhh0    = (const float*)d_in[4];
  const float* eb0      = (const float*)d_in[5];
  const float* eWih1    = (const float*)d_in[6];
  const float* eWhh1    = (const float*)d_in[7];
  const float* eb1      = (const float*)d_in[8];
  const float* fc1_w    = (const float*)d_in[9];
  const float* fc1_b    = (const float*)d_in[10];
  const float* fc2_w    = (const float*)d_in[11];
  const float* fc2_b    = (const float*)d_in[12];
  const float* dWih0    = (const float*)d_in[13];
  const float* dWhh0    = (const float*)d_in[14];
  const float* db0      = (const float*)d_in[15];
  const float* dWih1    = (const float*)d_in[16];
  const float* dWhh1    = (const float*)d_in[17];
  const float* db1      = (const float*)d_in[18];
  const float* out_w    = (const float*)d_in[19];
  const float* out_b    = (const float*)d_in[20];
  float* out = (float*)d_out;

  // ---- workspace carve-out ----
  char* w = (char*)d_ws;
  size_t off = 0;
  auto carve = [&](size_t bytes) -> void* {
    void* p = w + off;
    off = (off + bytes + 255) & ~(size_t)255;
    return p;
  };
  unsigned short* xe     = (unsigned short*)carve((size_t)TX_ * E_ * 2);
  unsigned short* Wih0b  = (unsigned short*)carve((size_t)G4_ * E_ * 2);
  unsigned short* Wih1b  = (unsigned short*)carve((size_t)G4_ * H_ * 2);
  float*          Gbuf   = (float*)carve((size_t)TX_ * G4_ * 4);
  unsigned short* ys0    = (unsigned short*)carve((size_t)TX_ * H_ * 2);
  unsigned short* Whh0b  = (unsigned short*)carve((size_t)G4_ * H_ * 2);
  unsigned short* Whh1b  = (unsigned short*)carve((size_t)G4_ * H_ * 2);
  unsigned short* dWhh0b = (unsigned short*)carve((size_t)G4_ * H_ * 2);
  unsigned short* dWih1b = (unsigned short*)carve((size_t)G4_ * H_ * 2);
  unsigned short* dWhh1b = (unsigned short*)carve((size_t)G4_ * H_ * 2);
  unsigned short* outWb  = (unsigned short*)carve((size_t)V_ * H_ * 2);
  float*    hvec  = (float*)carve(H_ * 4);
  float*    gates = (float*)carve(G4_ * 4);
  float*    h0b   = (float*)carve(H_ * 4);
  float*    c0b   = (float*)carve(H_ * 4);
  float*    h1b   = (float*)carve(H_ * 4);
  float*    c1b   = (float*)carve(H_ * 4);
  float*    red   = (float*)carve(64 * 3 * 4);
  int*      redi  = (int*)carve(64 * 4);
  float*    pub   = (float*)carve(2 * 4);
  unsigned* syn   = (unsigned*)carve(2 * 4);

  const int T = 256;
  auto blks = [&](long long n) { return (int)((n + T - 1) / T); };

  k_zero_u32<<<1, 32, 0, stream>>>(syn, 2);

  // weight conversions (fp32 -> bf16, L2-resident working set; all row-major)
  k_cvt<<<blks((long long)G4_ * E_), T, 0, stream>>>(eWih0, Wih0b, G4_ * E_);
  k_cvt<<<blks((long long)G4_ * H_), T, 0, stream>>>(eWih1, Wih1b, G4_ * H_);
  k_cvt<<<blks((long long)G4_ * H_), T, 0, stream>>>(eWhh0, Whh0b, G4_ * H_);
  k_cvt<<<blks((long long)G4_ * H_), T, 0, stream>>>(eWhh1, Whh1b, G4_ * H_);
  k_cvt<<<blks((long long)G4_ * H_), T, 0, stream>>>(dWhh0, dWhh0b, G4_ * H_);
  k_cvt<<<blks((long long)G4_ * H_), T, 0, stream>>>(dWih1, dWih1b, G4_ * H_);
  k_cvt<<<blks((long long)G4_ * H_), T, 0, stream>>>(dWhh1, dWhh1b, G4_ * H_);
  k_cvt<<<blks((long long)V_ * H_),  T, 0, stream>>>(out_w, outWb, V_ * H_);
  k_embed<<<(TX_ * E_) / T, T, 0, stream>>>(x, emb, xe);

  // ---- encoder ----
  // layer 0 input pre-projection: G = xe(2048x512) @ Wih0^T  (WMMA bf16)
  k_gemm_bf16<<<dim3(G4_ / 128, TX_ / 16), 256, 0, stream>>>(xe, Wih0b, Gbuf, TX_, G4_, E_);
  k_lstm_seq<<<32, 128, 0, stream>>>(Gbuf, Whh0b, eb0, hvec, gates, ys0, syn, syn + 1, TX_);
  // layer 1 input pre-projection: G = ys0(2048x1024) @ Wih1^T  (WMMA bf16)
  k_gemm_bf16<<<dim3(G4_ / 128, TX_ / 16), 256, 0, stream>>>(ys0, Wih1b, Gbuf, TX_, G4_, H_);
  k_lstm_seq<<<32, 128, 0, stream>>>(Gbuf, Whh1b, eb1, hvec, gates, nullptr, syn, syn + 1, TX_);

  // ---- bottleneck + decoder init ----
  k_fc<<<1, 1024, 0, stream>>>(fc1_w, fc1_b, fc2_w, fc2_b, hvec, h0b, c0b, h1b, c1b);
  k_row0<<<blks(V_), T, 0, stream>>>(out);

  // ---- autoregressive decoder (persistent, 511 steps) ----
  k_decoder<<<64, 256, 0, stream>>>(y, dWih0, db0, db1, dWhh0b, dWih1b, dWhh1b,
                                    outWb, out_b, h0b, c0b, h1b, c1b,
                                    gates, red, redi, pub, syn, syn + 1, out);
}
